// RGB_Depth_CrossMultiHeadAttention_67156108640529
// MI455X (gfx1250) — compile-verified, matrix operand order checked by reading, not run on hardware
//
#include <hip/hip_runtime.h>

// ---------------------------------------------------------------------------
// CDNA5 (gfx1250) fused RGB/Depth cross-modal MHA, bf16 WMMA pipeline
// with Tensor Data Mover (TDM) LDS staging.
// ---------------------------------------------------------------------------

typedef __attribute__((ext_vector_type(16))) __bf16 v16bf;
typedef __attribute__((ext_vector_type(8)))  __bf16 v8bf;
typedef __attribute__((ext_vector_type(8)))  float  v8f;
typedef __attribute__((ext_vector_type(4)))  unsigned int u32x4;
typedef __attribute__((ext_vector_type(8)))  int i32x8;
typedef __attribute__((ext_vector_type(4)))  int i32x4;

#define NB   8       // batch
#define NS   1024    // sequence
#define NH   12      // heads
#define ND   64      // head dim
#define HID  768
#define QKVN 2304    // 3*HID
#define ROWS2 16384  // 2 * NB * NS  (both modalities stacked)

__device__ __forceinline__ v8f wmma_bf16(v16bf a, v16bf b, v8f c) {
  // D = A(16x32) * B(32x16) + C, f32 accumulate
  return __builtin_amdgcn_wmma_f32_16x16x32_bf16(
      /*neg_a=*/false, a, /*neg_b=*/false, b,
      /*c_mod=*/(short)0, c, /*reuse_a=*/false, /*reuse_b=*/false);
}

// Build a 16-element bf16 fragment from two contiguous 8-element (16B) chunks.
__device__ __forceinline__ v16bf make_frag(const __bf16* p0, const __bf16* p1) {
  v8bf lo = *(const v8bf*)p0;
  v8bf hi = *(const v8bf*)p1;
  return __builtin_shufflevector(lo, hi, 0,1,2,3,4,5,6,7,8,9,10,11,12,13,14,15);
}

// ---------------------------------------------------------------------------
// TDM 2-D tile load: global (bf16, row stride stride0 elems) -> LDS, with
// optional hardware row padding.  D# layout per CDNA5 ISA ch.8.
//   pad_interval v: pad inserted after 2^(v+3) bytes
//   pad_amount  v: (v+1)*4 bytes of LDS padding
// 6-arg builtin flavor: (g0, g1, g2, g3, g_extra, cpol)
// ---------------------------------------------------------------------------
__device__ __forceinline__ void tdm_load_2d(unsigned int lds_off, const void* gptr,
                                            unsigned int tile_d0, unsigned int tile_d1,
                                            unsigned int tensor_d0, unsigned int tensor_d1,
                                            unsigned int stride0,
                                            unsigned int pad_interval,
                                            unsigned int pad_amount,
                                            unsigned int pad_en) {
  unsigned long long ga = (unsigned long long)(size_t)gptr;
  u32x4 g0;
  g0[0] = 1u;                                  // count=1, user descriptor
  g0[1] = lds_off;                             // LDS byte address
  g0[2] = (unsigned int)ga;                    // global_addr[31:0]
  g0[3] = (unsigned int)((ga >> 32) & 0x01FFFFFFu) | (2u << 30);  // addr[56:32] | type=2
  i32x8 g1;
  g1[0] = (int)((1u << 16) |                   // data_size = 1 (2-byte)
                (pad_en << 20) |               // pad_enable
                (pad_interval << 22) |
                (pad_amount << 25));
  g1[1] = (int)((tensor_d0 & 0xFFFFu) << 16);                       // tensor_dim0[15:0]
  g1[2] = (int)((tensor_d0 >> 16) | ((tensor_d1 & 0xFFFFu) << 16)); // d0[31:16], d1[15:0]
  g1[3] = (int)((tensor_d1 >> 16) | (tile_d0 << 16));               // d1[31:16], tile_dim0
  g1[4] = (int)(tile_d1 & 0xFFFFu);                                 // tile_dim1 (tile_dim2=0)
  g1[5] = (int)stride0;                                             // tensor_dim0_stride[31:0]
  g1[6] = 0;
  g1[7] = 0;
  i32x4 z4 = {0, 0, 0, 0};
  i32x8 z8 = {0, 0, 0, 0, 0, 0, 0, 0};
  __builtin_amdgcn_tensor_load_to_lds(g0, g1, z4, z4, z8, 0);
}

// ---------------------------------------------------------------------------
// Kernel 1: fp32 -> bf16 cast (grid-stride)
// ---------------------------------------------------------------------------
__global__ void cast_f32_bf16(const float* __restrict__ src,
                              __bf16* __restrict__ dst, int n) {
  int i = blockIdx.x * blockDim.x + threadIdx.x;
  int stride = gridDim.x * blockDim.x;
  for (; i < n; i += stride) dst[i] = (__bf16)src[i];
}

// ---------------------------------------------------------------------------
// Kernel 2/4: bf16 GEMM  C[M,N] = A[M,K] * W[K,N] + bias, 128x128x32 tiles.
// A tiles staged by TDM (double-buffered, hardware-padded rows), B staged
// transposed by the waves.  8 waves -> 4x2, each wave 2x4 WMMA tiles.
// ---------------------------------------------------------------------------
template <bool OUT_BF16>
__global__ __launch_bounds__(256)
void gemm_bf16(const __bf16* __restrict__ A, const __bf16* __restrict__ W,
               const float* __restrict__ bias, void* __restrict__ out,
               int M, int N, int K) {
  constexpr int BM = 128, BN = 128, BK = 32;
  constexpr int LDA = BK + 8;  // 40 bf16 = 80B rows (TDM pad reproduces this)
  constexpr int LDB = BK + 8;  // Bs stored transposed [n][k]
  __shared__ __bf16 As[2][BM * LDA];
  __shared__ __bf16 Bs[BN * LDB];

  const int tid   = threadIdx.x;
  const int wave  = tid >> 5;
  const int lane  = tid & 31;
  const int lrow  = lane & 15;
  const int khalf = lane >> 4;
  const int wm    = (wave >> 1) * 32;
  const int wn    = (wave & 1) * 64;
  const int m0    = blockIdx.y * BM;
  const int n0    = blockIdx.x * BN;

  v8f acc[2][4];
#pragma unroll
  for (int i = 0; i < 2; ++i)
#pragma unroll
    for (int j = 0; j < 4; ++j) acc[i][j] = {};

  // Prologue: TDM the first A tile into buffer 0 (wave 0 owns the TENSORcnt).
  if (wave == 0)
    tdm_load_2d((unsigned int)(size_t)&As[0][0], A + (size_t)m0 * K,
                /*tile*/ BK, BM, /*tensor*/ (unsigned)K, (unsigned)M,
                /*stride0*/ (unsigned)K, /*pad_interval(64B)*/ 3,
                /*pad_amount(16B)*/ 3, 1);

  int ab = 0;
  for (int k0 = 0; k0 < K; k0 += BK) {
    __syncthreads();  // previous iteration finished reading Bs / As
    // --- stage B tile transposed: Bs[n][k] ---
    {
      const int kr = tid >> 3;
      const int ns = (tid & 7) * 16;
      const __bf16* src = W + (size_t)(k0 + kr) * N + n0 + ns;
      if (k0 + BK < K) __builtin_prefetch(src + (size_t)BK * N, 0, 1);
#pragma unroll
      for (int i = 0; i < 16; ++i) Bs[(ns + i) * LDB + kr] = src[i];
    }
    if (wave == 0) __builtin_amdgcn_s_wait_tensorcnt(0);  // A tile landed
    __syncthreads();  // Bs staged && As[ab] ready for everyone
    // Prefetch next A tile into the other buffer while we compute.
    if (wave == 0 && k0 + BK < K)
      tdm_load_2d((unsigned int)(size_t)&As[ab ^ 1][0],
                  A + (size_t)m0 * K + k0 + BK,
                  BK, BM, (unsigned)K, (unsigned)M, (unsigned)K, 3, 3, 1);

    v16bf afr[2], bfr[4];
#pragma unroll
    for (int i = 0; i < 2; ++i) {
      const __bf16* p = &As[ab][(wm + i * 16 + lrow) * LDA + khalf * 8];
      afr[i] = make_frag(p, p + 16);
    }
#pragma unroll
    for (int j = 0; j < 4; ++j) {
      const __bf16* p = &Bs[(wn + j * 16 + lrow) * LDB + khalf * 8];
      bfr[j] = make_frag(p, p + 16);
    }
#pragma unroll
    for (int i = 0; i < 2; ++i)
#pragma unroll
      for (int j = 0; j < 4; ++j) acc[i][j] = wmma_bf16(afr[i], bfr[j], acc[i][j]);
    ab ^= 1;
  }

  // --- epilogue: C layout = lane -> col n, VGPR e -> row (khalf*8 + e) ---
#pragma unroll
  for (int i = 0; i < 2; ++i)
#pragma unroll
    for (int j = 0; j < 4; ++j) {
      const int gn  = n0 + wn + j * 16 + lrow;
      const float bv = bias[gn];
#pragma unroll
      for (int e = 0; e < 8; ++e) {
        const int gm = m0 + wm + i * 16 + khalf * 8 + e;
        const float v = acc[i][j][e] + bv;
        if constexpr (OUT_BF16)
          ((__bf16*)out)[(size_t)gm * N + gn] = (__bf16)v;
        else
          ((float*)out)[(size_t)gm * N + gn] = v;
      }
    }
}

// ---------------------------------------------------------------------------
// Kernel 3: fused dual-modality attention.
// One block = (b, h, 16 q-rows). fp32 score slabs for both modalities live in
// LDS (CDNA5 320KB WGP LDS), cross-modal mix + dual softmax in place, then
// P@V WMMA with V chunks staged by TDM (double-buffered, padded rows).
// ---------------------------------------------------------------------------
__global__ __launch_bounds__(256)
void dual_attn(const __bf16* __restrict__ qkv, __bf16* __restrict__ obuf) {
  constexpr int LDS_S = NS + 4;   // 1028 f32 row stride
  constexpr int VCH   = 128;      // V chunk rows
  constexpr int LDV   = ND + 8;   // 72 bf16 row stride (TDM pad reproduces)
  __shared__ float  Sc[2][16 * LDS_S];       // 2 * 65,792 B
  __shared__ __bf16 Vs[2][2][VCH * LDV];     // [buf][mod], 4 * 18,432 B

  const int tid   = threadIdx.x;
  const int wave  = tid >> 5;
  const int lane  = tid & 31;
  const int lrow  = lane & 15;
  const int khalf = lane >> 4;

  const int qt = blockIdx.x & 63;        // q tile (S/16 = 64)
  const int bh = blockIdx.x >> 6;
  const int b  = bh / NH;
  const int h  = bh % NH;

  // ---- Phase 1: score slabs  S[mod][16][wave*128 .. +128) ----
  v16bf qf[2][2];
#pragma unroll
  for (int mod = 0; mod < 2; ++mod)
#pragma unroll
    for (int ks = 0; ks < 2; ++ks) {
      const size_t row = (size_t)(mod * NB + b) * NS + qt * 16 + lrow;
      const __bf16* p = qkv + row * QKVN + h * ND + ks * 32 + khalf * 8;
      qf[mod][ks] = make_frag(p, p + 16);
    }

  const int colBase = wave * 128;
#pragma unroll
  for (int mod = 0; mod < 2; ++mod) {
    for (int st = 0; st < 8; ++st) {
      const int sbase = colBase + st * 16;
      const size_t krow = (size_t)(mod * NB + b) * NS + sbase + lrow;
      const __bf16* kp = qkv + krow * QKVN + HID + h * ND;  // K^T as B matrix
      v16bf kf0 = make_frag(kp + khalf * 8,      kp + khalf * 8 + 16);
      v16bf kf1 = make_frag(kp + 32 + khalf * 8, kp + 32 + khalf * 8 + 16);
      v8f acc = {};
      acc = wmma_bf16(qf[mod][0], kf0, acc);
      acc = wmma_bf16(qf[mod][1], kf1, acc);
#pragma unroll
      for (int e = 0; e < 8; ++e)
        Sc[mod][(khalf * 8 + e) * LDS_S + sbase + lrow] = acc[e];
    }
  }
  __syncthreads();

  // ---- Phase 2: cross-modal mix + dual softmax (row r owned by 16 lanes) ----
  {
    const int r  = tid >> 4;
    const int c0 = tid & 15;
    float* si = &Sc[0][r * LDS_S];
    float* sd = &Sc[1][r * LDS_S];
    const float sc = 0.125f;  // 1/sqrt(64)
    float mi = -3.0e30f, md = -3.0e30f;
    for (int j = 0; j < NS / 16; ++j) {
      const int c = c0 + j * 16;
      const float a = si[c], d = sd[c];
      const float xi = sc * (a + 0.5f * d);   // ALPHA = 0.5
      const float xd = sc * (d + 0.5f * a);   // BETA  = 0.5
      si[c] = xi; sd[c] = xd;
      mi = fmaxf(mi, xi); md = fmaxf(md, xd);
    }
#pragma unroll
    for (int o = 8; o >= 1; o >>= 1) {
      mi = fmaxf(mi, __shfl_xor(mi, o, 16));
      md = fmaxf(md, __shfl_xor(md, o, 16));
    }
    float sumi = 0.f, sumd = 0.f;
    for (int j = 0; j < NS / 16; ++j) {
      const int c = c0 + j * 16;
      const float ei = __expf(si[c] - mi);
      const float ed = __expf(sd[c] - md);
      si[c] = ei; sd[c] = ed;
      sumi += ei; sumd += ed;
    }
#pragma unroll
    for (int o = 8; o >= 1; o >>= 1) {
      sumi += __shfl_xor(sumi, o, 16);
      sumd += __shfl_xor(sumd, o, 16);
    }
    const float ri = 1.0f / sumi, rd = 1.0f / sumd;
    for (int j = 0; j < NS / 16; ++j) {
      const int c = c0 + j * 16;
      si[c] *= ri; sd[c] *= rd;
    }
  }
  __syncthreads();

  // ---- Phase 3: O = P @ V.  wave -> (modality, 16-col tile of D). ----
  const int mod = wave >> 2;
  const int nt  = (wave & 3) * 16;
  v8f oacc = {};

  // TDM issue for one V chunk (both modalities) into buffer `buf`.
  auto issueV = [&](int ch, int buf) {
#pragma unroll
    for (int m2 = 0; m2 < 2; ++m2) {
      const size_t vrow0 = (size_t)(m2 * NB + b) * NS + (size_t)ch * VCH;
      tdm_load_2d((unsigned int)(size_t)&Vs[buf][m2][0],
                  qkv + vrow0 * QKVN + 2 * HID + h * ND,
                  /*tile*/ ND, VCH, /*tensor*/ ND, VCH,
                  /*stride0*/ QKVN, /*pad_interval(128B)*/ 4,
                  /*pad_amount(16B)*/ 3, 1);
    }
  };

  if (wave == 0) issueV(0, 0);
  int vb = 0;
  for (int ch = 0; ch < NS / VCH; ++ch) {
    if (wave == 0) __builtin_amdgcn_s_wait_tensorcnt(0);
    __syncthreads();                        // Vs[vb] landed; prev buf free
    if (wave == 0 && ch + 1 < NS / VCH) issueV(ch + 1, vb ^ 1);
#pragma unroll
    for (int ks = 0; ks < VCH / 32; ++ks) {
      const int kb = ks * 32;
      // A fragment: probabilities from LDS, f32 -> bf16 on the fly
      v16bf af;
      {
        const float* p = &Sc[mod][lrow * LDS_S + ch * VCH + kb + khalf * 8];
#pragma unroll
        for (int e = 0; e < 8; ++e) af[e]     = (__bf16)p[e];
#pragma unroll
        for (int e = 0; e < 8; ++e) af[8 + e] = (__bf16)p[16 + e];
      }
      // B fragment: V rows (k = s) striped from LDS
      v16bf bfv;
      {
        const __bf16* vp = &Vs[vb][mod][0];
        const int n = nt + lrow;
#pragma unroll
        for (int e = 0; e < 8; ++e) bfv[e]     = vp[(kb + khalf * 8 + e) * LDV + n];
#pragma unroll
        for (int e = 0; e < 8; ++e) bfv[8 + e] = vp[(kb + 16 + khalf * 8 + e) * LDV + n];
      }
      oacc = wmma_bf16(af, bfv, oacc);
    }
    vb ^= 1;
  }
  // write O tile -> [mod][B][S][H*D] bf16
  {
    const int gn = h * ND + nt + lrow;
#pragma unroll
    for (int e = 0; e < 8; ++e) {
      const size_t gm = (size_t)(mod * NB + b) * NS + qt * 16 + khalf * 8 + e;
      obuf[gm * HID + gn] = (__bf16)oacc[e];
    }
  }
}

// ---------------------------------------------------------------------------
// Launch
// ---------------------------------------------------------------------------
extern "C" void kernel_launch(void* const* d_in, const int* in_sizes, int n_in,
                              void* d_out, int out_size, void* d_ws, size_t ws_size,
                              hipStream_t stream) {
  const float* img   = (const float*)d_in[0];
  const float* dpt   = (const float*)d_in[1];
  const float* w_qkv = (const float*)d_in[2];
  const float* b_qkv = (const float*)d_in[3];
  const float* w_out = (const float*)d_in[4];
  const float* b_out = (const float*)d_in[5];
  float* out = (float*)d_out;

  // Workspace layout (all 256B aligned by construction)
  char* ws = (char*)d_ws;
  const size_t BS = (size_t)NB * NS;                 // 8192 rows per modality
  __bf16* Xbf   = (__bf16*)(ws);                                   // 16384 x 768
  __bf16* Wqkvb = (__bf16*)(ws + 2 * BS * HID * 2);                // 768 x 2304
  __bf16* Woutb = (__bf16*)((char*)Wqkvb + (size_t)HID * QKVN * 2);// 768 x 768
  __bf16* QKV   = (__bf16*)((char*)Woutb + (size_t)HID * HID * 2); // 16384 x 2304
  __bf16* Obuf  = (__bf16*)((char*)QKV + 2 * BS * QKVN * 2);       // 16384 x 768

  // 1) fp32 -> bf16 casts
  cast_f32_bf16<<<1024, 256, 0, stream>>>(img, Xbf, (int)(BS * HID));
  cast_f32_bf16<<<1024, 256, 0, stream>>>(dpt, Xbf + BS * HID, (int)(BS * HID));
  cast_f32_bf16<<<512, 256, 0, stream>>>(w_qkv, Wqkvb, HID * QKVN);
  cast_f32_bf16<<<256, 256, 0, stream>>>(w_out, Woutb, HID * HID);

  // 2) QKV projection for both modalities: [16384,768] x [768,2304] + b_qkv
  {
    dim3 grid(QKVN / 128, ROWS2 / 128);  // (18, 128)
    gemm_bf16<true><<<grid, 256, 0, stream>>>(Xbf, Wqkvb, b_qkv, QKV,
                                              ROWS2, QKVN, HID);
  }

  // 3) fused dual-modality attention: one block per (b, h, 16 q-rows)
  dual_attn<<<NB * NH * (NS / 16), 256, 0, stream>>>(QKV, Obuf);

  // 4) output projection straight into d_out (img half, then dpt half)
  {
    dim3 grid(HID / 128, ROWS2 / 128);   // (6, 128)
    gemm_bf16<false><<<grid, 256, 0, stream>>>(Obuf, Woutb, b_out, out,
                                               ROWS2, HID, HID);
  }
}
